// DeformableShiftMLP_75170517614823
// MI455X (gfx1250) — matrix operand order repaired and checked by reading, not compiled, not run on hardware
//
#include <hip/hip_runtime.h>
#include <hip/hip_bf16.h>

// ---------------------------------------------------------------------------
// Types for CDNA5 WMMA (wave32): v_wmma_f32_16x16x32_bf16
// ---------------------------------------------------------------------------
typedef __attribute__((ext_vector_type(16))) __bf16 bf16x16;
typedef __attribute__((ext_vector_type(8)))  __bf16 bf16x8;
typedef __attribute__((ext_vector_type(8)))  float  f32x8;

__device__ __forceinline__ unsigned short f32_to_bf16(float f) {
  union { float f; unsigned int u; } v; v.f = f;
  unsigned int u = v.u;
  unsigned int lsb = (u >> 16) & 1u;
  u += 0x7fffu + lsb;                 // round-to-nearest-even
  return (unsigned short)(u >> 16);
}

// ---------------------------------------------------------------------------
// Kernel 1: gated soft-shift along H on C=96 channels.
// One wave (32 lanes) per token; 3 channels per lane (stride-32, coalesced).
// Output: bf16 [B*N, 96] row-major (A matrix for fc1 WMMA GEMM).
// ---------------------------------------------------------------------------
__global__ void shift_h_kernel(const float* __restrict__ x,     // [B,N,96]
                               const float* __restrict__ ghw,   // [5,96]
                               const float* __restrict__ ghb,   // [5]
                               unsigned short* __restrict__ out,
                               int B, int H, int W) {
  const int C = 96;
  const int lane = threadIdx.x & 31;
  const int wave = threadIdx.x >> 5;
  const size_t token = (size_t)blockIdx.x * (blockDim.x >> 5) + wave;
  const int N = H * W;
  if (token >= (size_t)B * N) return;
  const int b = (int)(token / N);
  const int n = (int)(token % N);
  const int h = n / W, w = n % W;

  const float* xt = x + token * C;
  const float v0 = xt[lane], v1 = xt[lane + 32], v2 = xt[lane + 64];

  float lg[5];
#pragma unroll
  for (int s = 0; s < 5; ++s) {
    float p = v0 * ghw[s * C + lane] + v1 * ghw[s * C + lane + 32] +
              v2 * ghw[s * C + lane + 64];
#pragma unroll
    for (int off = 16; off > 0; off >>= 1) p += __shfl_xor(p, off, 32);
    lg[s] = p + ghb[s];
  }
  float mx = lg[0];
#pragma unroll
  for (int s = 1; s < 5; ++s) mx = fmaxf(mx, lg[s]);
  float e[5], sum = 0.f;
#pragma unroll
  for (int s = 0; s < 5; ++s) { e[s] = __expf(lg[s] - mx); sum += e[s]; }
  const float inv = 1.f / sum;

  float a0 = 0.f, a1 = 0.f, a2 = 0.f;
#pragma unroll
  for (int i = 0; i < 5; ++i) {
    const int hh = h - (i - 2);
    if (hh < 0 || hh >= H) continue;
    const float wt = e[i] * inv;
    const float* xs = x + ((size_t)b * N + (size_t)hh * W + w) * C;
    a0 += xs[lane] * wt; a1 += xs[lane + 32] * wt; a2 += xs[lane + 64] * wt;
  }
  unsigned short* ot = out + token * C;
  ot[lane]      = f32_to_bf16(a0);
  ot[lane + 32] = f32_to_bf16(a1);
  ot[lane + 64] = f32_to_bf16(a2);
}

// ---------------------------------------------------------------------------
// Kernel 2: pack f32 weight [K,N] into bf16 WMMA B-fragment tiles.
// Tile (kb,nb) holds a 32x16 K-major fragment; lane L owns column N=nb*16+(L&15),
// K rows (L>=16?16:0)+j, j=0..15, stored contiguously (32 B per lane).
// ---------------------------------------------------------------------------
__global__ void pack_b_kernel(const float* __restrict__ Bsrc,   // [K,N]
                              unsigned short* __restrict__ Bp,
                              int K, int N) {
  const int idx = blockIdx.x * blockDim.x + threadIdx.x;
  if (idx >= K * N) return;
  const int j    = idx & 15;
  const int lane = (idx >> 4) & 31;
  const int tile = idx >> 9;
  const int KB = K >> 5;
  const int kb = tile % KB;
  const int nb = tile / KB;
  const int kk = kb * 32 + ((lane >= 16) ? 16 : 0) + j;
  const int nn = nb * 16 + (lane & 15);
  Bp[idx] = f32_to_bf16(Bsrc[(size_t)kk * N + nn]);
}

// ---------------------------------------------------------------------------
// Kernel 3: GEMM  C[M,N] = A[M,K](bf16) * B[K,N](packed bf16) + bias, f32 out.
// 8 waves / 256-thread block; each wave computes a 16 x (NT*16) strip with
// v_wmma_f32_16x16x32_bf16. Wide NT amortizes A traffic (B is L2-resident).
// ---------------------------------------------------------------------------
template <int NT>
__global__ void gemm_bf16_wmma(const unsigned short* __restrict__ A,
                               const unsigned short* __restrict__ Bp,
                               const float* __restrict__ bias,
                               float* __restrict__ C,
                               int M, int N, int K) {
  const int lane = threadIdx.x & 31;
  const int wave = threadIdx.x >> 5;
  const int m0 = (blockIdx.x * 8 + wave) * 16;
  const int n0 = blockIdx.y * (NT * 16);
  const int KB = K >> 5;

  f32x8 acc[NT];
#pragma unroll
  for (int t = 0; t < NT; ++t)
#pragma unroll
    for (int r = 0; r < 8; ++r) acc[t][r] = 0.f;

  const int mrow  = m0 + (lane & 15);
  const int kbase = (lane >> 4) * 16;         // 0 or 16

  for (int kb = 0; kb < KB; ++kb) {
    const unsigned short* ap = A + (size_t)mrow * K + kb * 32 + kbase;
    const bf16x8 alo = *(const bf16x8*)ap;          // K = kbase + 0..7
    const bf16x8 ahi = *(const bf16x8*)(ap + 8);    // K = kbase + 8..15
    bf16x16 a;
#pragma unroll
    for (int j = 0; j < 8; ++j) { a[j] = alo[j]; a[8 + j] = ahi[j]; }

#pragma unroll
    for (int t = 0; t < NT; ++t) {
      const unsigned short* bp =
          Bp + (((size_t)((n0 >> 4) + t) * KB + kb) * 32 + lane) * 16;
      const bf16x16 bfrag = *(const bf16x16*)bp;
      acc[t] = __builtin_amdgcn_wmma_f32_16x16x32_bf16(
          /*neg_a=*/false, a, /*neg_b=*/false, bfrag,
          /*c_mod=*/(short)0, acc[t], /*reuse_a=*/false, /*reuse_b=*/false);
    }
  }

  // C/D layout: VGPR r -> row m0 + r + (lane>=16 ? 8 : 0), col n0 + (lane&15)
  const int rbase = m0 + ((lane >> 4) * 8);
  const int cbase = n0 + (lane & 15);
#pragma unroll
  for (int t = 0; t < NT; ++t) {
    const int col = cbase + t * 16;
    const float bb = bias[col];
#pragma unroll
    for (int r = 0; r < 8; ++r)
      C[(size_t)(rbase + r) * N + col] = acc[t][r] + bb;
  }
}

// ---------------------------------------------------------------------------
// Kernel 4: depthwise 3x3 (NHWC) + bias + exact GELU, register sliding window.
// Thread owns one channel, computes TH output rows, slides along W keeping a
// (TH+2)-row x 3-column window: (TH+2)/TH input reads per output (vs 9).
// ---------------------------------------------------------------------------
template <int TH>
__global__ void dwconv_gelu_kernel(const float* __restrict__ in,  // [B,H,W,384]
                                   const float* __restrict__ wgt, // [384,1,3,3]
                                   const float* __restrict__ bias,
                                   float* __restrict__ out,
                                   int B, int H, int W) {
  const int C = 384;
  const int c = blockIdx.y * blockDim.x + threadIdx.x;     // channel
  const int rowsPer = H / TH;
  const int b  = blockIdx.x / rowsPer;
  const int h0 = (blockIdx.x % rowsPer) * TH;

  float k[9];
#pragma unroll
  for (int i = 0; i < 9; ++i) k[i] = wgt[c * 9 + i];
  const float bb = bias[c];

  const size_t rowStride = (size_t)W * C;
  const float* base = in + (size_t)b * H * rowStride + c;

  float col0[TH + 2], col1[TH + 2], col2[TH + 2];
#pragma unroll
  for (int r = 0; r < TH + 2; ++r) {
    const int hh = h0 - 1 + r;
    col0[r] = 0.f;                                          // w = -1 (pad)
    col1[r] = (hh >= 0 && hh < H) ? base[(size_t)hh * rowStride] : 0.f;
  }

  for (int w = 0; w < W; ++w) {
    const bool hasR = (w + 1 < W);
#pragma unroll
    for (int r = 0; r < TH + 2; ++r) {
      const int hh = h0 - 1 + r;
      col2[r] = (hasR && hh >= 0 && hh < H)
                    ? base[(size_t)hh * rowStride + (size_t)(w + 1) * C]
                    : 0.f;
    }
#pragma unroll
    for (int r = 0; r < TH; ++r) {
      float s = bb;
      s += col0[r]     * k[0] + col1[r]     * k[1] + col2[r]     * k[2];
      s += col0[r + 1] * k[3] + col1[r + 1] * k[4] + col2[r + 1] * k[5];
      s += col0[r + 2] * k[6] + col1[r + 2] * k[7] + col2[r + 2] * k[8];
      const float g = 0.5f * s * (1.0f + erff(s * 0.70710678118654752440f));
      out[((size_t)b * H + (h0 + r)) * rowStride + (size_t)w * C + c] = g;
    }
#pragma unroll
    for (int r = 0; r < TH + 2; ++r) { col0[r] = col1[r]; col1[r] = col2[r]; }
  }
}

// ---------------------------------------------------------------------------
// Kernel 5: gated soft-shift along W on Ch=384 channels (12 ch/lane).
// Output bf16 [B*N, 384] (A matrix for fc2 GEMM).
// ---------------------------------------------------------------------------
__global__ void shift_w_kernel(const float* __restrict__ x,     // [B,N,384]
                               const float* __restrict__ gww,   // [5,384]
                               const float* __restrict__ gwb,   // [5]
                               unsigned short* __restrict__ out,
                               int B, int H, int W) {
  const int C = 384;
  const int lane = threadIdx.x & 31;
  const int wave = threadIdx.x >> 5;
  const size_t token = (size_t)blockIdx.x * (blockDim.x >> 5) + wave;
  const int N = H * W;
  if (token >= (size_t)B * N) return;
  const int b = (int)(token / N);
  const int n = (int)(token % N);
  const int h = n / W, w = n % W;

  const float* xt = x + token * C;
  float v[12];
#pragma unroll
  for (int r = 0; r < 12; ++r) v[r] = xt[lane + 32 * r];

  float lg[5];
#pragma unroll
  for (int s = 0; s < 5; ++s) {
    float p = 0.f;
#pragma unroll
    for (int r = 0; r < 12; ++r) p += v[r] * gww[s * C + lane + 32 * r];
#pragma unroll
    for (int off = 16; off > 0; off >>= 1) p += __shfl_xor(p, off, 32);
    lg[s] = p + gwb[s];
  }
  float mx = lg[0];
#pragma unroll
  for (int s = 1; s < 5; ++s) mx = fmaxf(mx, lg[s]);
  float e[5], sum = 0.f;
#pragma unroll
  for (int s = 0; s < 5; ++s) { e[s] = __expf(lg[s] - mx); sum += e[s]; }
  const float inv = 1.f / sum;

  float acc[12];
#pragma unroll
  for (int r = 0; r < 12; ++r) acc[r] = 0.f;
#pragma unroll
  for (int i = 0; i < 5; ++i) {
    const int ww = w - (i - 2);
    if (ww < 0 || ww >= W) continue;
    const float wt = e[i] * inv;
    const float* xs = x + ((size_t)b * N + (size_t)h * W + ww) * C;
#pragma unroll
    for (int r = 0; r < 12; ++r) acc[r] += xs[lane + 32 * r] * wt;
  }
  unsigned short* ot = out + token * C;
#pragma unroll
  for (int r = 0; r < 12; ++r) ot[lane + 32 * r] = f32_to_bf16(acc[r]);
}

// ---------------------------------------------------------------------------
// Launcher
// ---------------------------------------------------------------------------
extern "C" void kernel_launch(void* const* d_in, const int* in_sizes, int n_in,
                              void* d_out, int out_size, void* d_ws,
                              size_t ws_size, hipStream_t stream) {
  (void)n_in; (void)out_size; (void)ws_size;
  const float* x     = (const float*)d_in[0];
  // d_in[1] = H, d_in[2] = W (device-resident scalars; values fixed by reference)
  const float* fc1_w = (const float*)d_in[3];
  const float* fc1_b = (const float*)d_in[4];
  const float* dw_w  = (const float*)d_in[5];
  const float* dw_b  = (const float*)d_in[6];
  const float* fc2_w = (const float*)d_in[7];
  const float* fc2_b = (const float*)d_in[8];
  const float* gh_w  = (const float*)d_in[9];
  const float* gh_b  = (const float*)d_in[10];
  const float* gw_w  = (const float*)d_in[11];
  const float* gw_b  = (const float*)d_in[12];

  const int H = 128, W = 128, C = 96, Ch = 384;
  const int N = H * W;
  const int B = in_sizes[0] / (N * C);
  const int M = B * N;                       // 262144 rows

  char* ws = (char*)d_ws;
  size_t off = 0;
  unsigned short* mixedH = (unsigned short*)(ws + off); off += (size_t)M * C * 2;
  const size_t fc1out_off = off;
  float* fc1out          = (float*)(ws + off); off += (size_t)M * Ch * 4;
  float* geluout         = (float*)(ws + off); off += (size_t)M * Ch * 4;
  unsigned short* mixedW = (unsigned short*)(ws + fc1out_off);  // reuse fc1out
  unsigned short* pB1    = (unsigned short*)(ws + off); off += (size_t)C * Ch * 2;
  unsigned short* pB2    = (unsigned short*)(ws + off); off += (size_t)Ch * C * 2;

  // Pack GEMM weights into WMMA B-fragment layout (bf16).
  pack_b_kernel<<<(C * Ch + 255) / 256, 256, 0, stream>>>(fc1_w, pB1, C, Ch);
  pack_b_kernel<<<(Ch * C + 255) / 256, 256, 0, stream>>>(fc2_w, pB2, Ch, C);

  // 1) soft-shift along H (one wave per token).
  shift_h_kernel<<<M / 8, 256, 0, stream>>>(x, gh_w, gh_b, mixedH, B, H, W);

  // 2) fc1: [M,96] x [96,384] + bias  (bf16 WMMA, f32 accumulate).
  //    NT=8 -> each wave does 16x128; A re-read only 3x.
  gemm_bf16_wmma<8><<<dim3(M / 128, Ch / 128), 256, 0, stream>>>(
      mixedH, pB1, fc1_b, fc1out, M, Ch, C);

  // 3) depthwise 3x3 + bias + exact GELU (register sliding window, TH=4).
  dwconv_gelu_kernel<4><<<dim3(B * (H / 4), Ch / 128), 128, 0, stream>>>(
      fc1out, dw_w, dw_b, geluout, B, H, W);

  // 4) soft-shift along W (one wave per token, 384 channels).
  shift_w_kernel<<<M / 8, 256, 0, stream>>>(geluout, gw_w, gw_b, mixedW, B, H, W);

  // 5) fc2: [M,384] x [384,96] + bias -> d_out (f32).
  //    NT=6 -> each wave does 16x96; A read exactly once.
  gemm_bf16_wmma<6><<<dim3(M / 128, 1), 256, 0, stream>>>(
      mixedW, pB2, fc2_b, (float*)d_out, M, C, Ch);
}
